// CausalSelfAttention_609885356942
// MI455X (gfx1250) — matrix operand order repaired and checked by reading, not compiled, not run on hardware
//
#include <hip/hip_runtime.h>

// ---------------------------------------------------------------------------
// Causal self-attention for MI455X (gfx1250, wave32, WMMA 16x16x32 f16).
// B=2 T=2048 C=1024 H=16 D=64.  All matmuls on v_wmma_f32_16x16x32_f16.
// GEMM waves compute 32x64 tiles (8 WMMA per k-step, B-frags reused 2x).
// ---------------------------------------------------------------------------

typedef __attribute__((ext_vector_type(16))) _Float16 v16h;
typedef __attribute__((ext_vector_type(8)))  _Float16 v8h;
typedef __attribute__((ext_vector_type(8)))  float    v8f;

#define BB 2
#define TT 2048
#define CC 1024
#define HH 16
#define DD 64

// ---- WMMA fragment helpers (layouts per CDNA5 ISA 7.12.2) ------------------
// A 16x32 f16: lane L holds row M=L&15, K chunks {half*8..+7, 16+half*8..+7}
__device__ __forceinline__ v16h load_frag_a(const _Float16* A, int ld, int lane) {
  const int m = lane & 15, half = lane >> 4;
  const _Float16* p = A + (size_t)m * ld + half * 8;
  v8h lo = *(const v8h*)(p);
  v8h hi = *(const v8h*)(p + 16);
  v16h r;
#pragma unroll
  for (int i = 0; i < 8; ++i) { r[i] = lo[i]; r[8 + i] = hi[i]; }
  return r;
}

// B 32x16 f16, operand given transposed (Bt[n][k], n = output column):
// lane L holds column N=L&15, K chunks {half*8..+7, 16+half*8..+7} (matches A).
__device__ __forceinline__ v16h load_frag_b(const _Float16* Bt, int ld, int lane) {
  const int n = lane & 15, half = lane >> 4;
  const _Float16* p = Bt + (size_t)n * ld + half * 8;
  v8h lo = *(const v8h*)(p);
  v8h hi = *(const v8h*)(p + 16);
  v16h r;
#pragma unroll
  for (int i = 0; i < 8; ++i) { r[i] = lo[i]; r[8 + i] = hi[i]; }
  return r;
}

__device__ __forceinline__ v8f wmma16(v16h a, v16h b, v8f c) {
  return __builtin_amdgcn_wmma_f32_16x16x32_f16(false, a, false, b, (short)0, c,
                                                false, false);
}

// ---- conversion kernels ----------------------------------------------------
__global__ __launch_bounds__(256) void cvt_f32_to_f16(const float* __restrict__ s,
                                                      _Float16* __restrict__ d, int n) {
  int i = blockIdx.x * 256 + threadIdx.x;
  if (i < n) d[i] = (_Float16)s[i];
}

// src [rows][cols] f32 -> dst [cols][rows] f16 (transpose), 16x16 thread tiles
__global__ __launch_bounds__(256) void cvt_transpose(const float* __restrict__ s,
                                                     _Float16* __restrict__ d,
                                                     int rows, int cols) {
  const int c = blockIdx.x * 16 + (threadIdx.x & 15);
  const int r = blockIdx.y * 16 + (threadIdx.x >> 4);
  d[(size_t)c * rows + r] = (_Float16)s[(size_t)r * cols + c];
}

// ---- QKV projection: [4096,1024]f16 @ [1024,3072] + bias -------------------
// Wave computes 32(M)x64(N); scatters q (scaled 1/sqrt(D)) / k to [B,H,T,D],
// v transposed to [B,H,D,T].  Region routing is wave-uniform (64-aligned).
__global__ __launch_bounds__(256) void gemm_qkv(const _Float16* __restrict__ Ah,
                                                const _Float16* __restrict__ Bt,
                                                const float* __restrict__ bias,
                                                _Float16* __restrict__ qh,
                                                _Float16* __restrict__ kh,
                                                _Float16* __restrict__ vt) {
  const int lane = threadIdx.x & 31;
  const int gw = blockIdx.x * 8 + (threadIdx.x >> 5);
  const int nt = gw % (3 * CC / 64);       // 48 N-groups of 64
  const int mt = gw / (3 * CC / 64);       // 128 M-tiles of 32
  const int m0 = mt * 32, n0 = nt * 64;
  const int n = lane & 15, half = lane >> 4;

  v8f acc[2][4];
#pragma unroll
  for (int s = 0; s < 4; ++s) {
    float bv = bias[n0 + s * 16 + n];
#pragma unroll
    for (int v = 0; v < 8; ++v) { acc[0][s][v] = bv; acc[1][s][v] = bv; }
  }

  const _Float16* Ap0 = Ah + (size_t)m0 * CC;
  const _Float16* Ap1 = Ap0 + (size_t)16 * CC;
  for (int k = 0; k < CC; k += 32) {
    // hoist all fragment loads -> one clause + staggered waits, then 8 WMMAs
    v16h af0 = load_frag_a(Ap0 + k, CC, lane);
    v16h af1 = load_frag_a(Ap1 + k, CC, lane);
    v16h bf[4];
#pragma unroll
    for (int s = 0; s < 4; ++s)
      bf[s] = load_frag_b(Bt + (size_t)(n0 + s * 16) * CC + k, CC, lane);
#pragma unroll
    for (int s = 0; s < 4; ++s) acc[0][s] = wmma16(af0, bf[s], acc[0][s]);
#pragma unroll
    for (int s = 0; s < 4; ++s) acc[1][s] = wmma16(af1, bf[s], acc[1][s]);
  }

  // ---- scatter epilogue -----------------------------------------------------
  // Region (q/k/v) and batch index are wave-uniform: hoist them out of the
  // store loops and make uniformity explicit via readfirstlane -> scalar branch.
  const int region = __builtin_amdgcn_readfirstlane(nt >> 4);     // 0=Q 1=K 2=V
  const int b      = __builtin_amdgcn_readfirstlane(m0 >> 11);    // batch
  const int t0     = __builtin_amdgcn_readfirstlane(m0 & (TT - 1));
  const int c0     = n0 - region * CC;       // column within the 1024-wide region

  if (region == 0) {            // Q: [B,H,T,D], pre-scaled by 1/sqrt(D)
#pragma unroll
    for (int s = 0; s < 4; ++s) {
      const int cq = c0 + s * 16 + n, h = cq >> 6, d = cq & 63;
      _Float16* dst = qh + (((size_t)(b * HH + h)) * TT + t0) * DD + d;
#pragma unroll
      for (int g = 0; g < 2; ++g)
#pragma unroll
        for (int v = 0; v < 8; ++v)
          dst[(size_t)(g * 16 + v + 8 * half) * DD] = (_Float16)(acc[g][s][v] * 0.125f);
    }
  } else if (region == 1) {     // K: [B,H,T,D]
#pragma unroll
    for (int s = 0; s < 4; ++s) {
      const int cq = c0 + s * 16 + n, h = cq >> 6, d = cq & 63;
      _Float16* dst = kh + (((size_t)(b * HH + h)) * TT + t0) * DD + d;
#pragma unroll
      for (int g = 0; g < 2; ++g)
#pragma unroll
        for (int v = 0; v < 8; ++v)
          dst[(size_t)(g * 16 + v + 8 * half) * DD] = (_Float16)acc[g][s][v];
    }
  } else {                      // V transposed: [B,H,D,T]
#pragma unroll
    for (int s = 0; s < 4; ++s) {
      const int cq = c0 + s * 16 + n, h = cq >> 6, d = cq & 63;
      _Float16* dst = vt + (((size_t)(b * HH + h)) * DD + d) * TT + t0;
#pragma unroll
      for (int g = 0; g < 2; ++g)
#pragma unroll
        for (int v = 0; v < 8; ++v)
          dst[g * 16 + v + 8 * half] = (_Float16)acc[g][s][v];
    }
  }
}

// ---- flash attention: one wave per (b,h, 16-query tile) --------------------
__global__ __launch_bounds__(256) void attn_kernel(const _Float16* __restrict__ qh,
                                                   const _Float16* __restrict__ kh,
                                                   const _Float16* __restrict__ vt,
                                                   _Float16* __restrict__ yh) {
  __shared__ __align__(16) _Float16 plds_all[8][16 * 32];
  const int lane = threadIdx.x & 31;
  const int wid = threadIdx.x >> 5;
  _Float16* plds = plds_all[wid];

  const int gw = blockIdx.x * 8 + wid;
  const int qt = gw & 127;                 // 128 query tiles
  const int bh = gw >> 7;                  // 0..31
  const int b = bh >> 4, h = bh & 15;
  const int q0 = qt * 16;
  const int n = lane & 15, half = lane >> 4;

  const _Float16* Q = qh + ((size_t)bh * TT + q0) * DD;
  const _Float16* K = kh + (size_t)bh * TT * DD;
  const _Float16* V = vt + (size_t)bh * DD * TT;

  const v16h qf0 = load_frag_a(Q, DD, lane);        // head dims 0..31
  const v16h qf1 = load_frag_a(Q + 32, DD, lane);   // head dims 32..63

  float mrow[8], lrow[8];
  v8f o[4] = {v8f{}, v8f{}, v8f{}, v8f{}};
#pragma unroll
  for (int v = 0; v < 8; ++v) { mrow[v] = -__builtin_inff(); lrow[v] = 0.f; }

  const int nkb = (q0 + 16 + 31) >> 5;     // causal bound, 32 keys per block
  for (int kb = 0; kb < nkb; ++kb) {
    const int k0 = kb * 32;

    // prefetch next key block (K and V tiles) into cache while we compute
    if (kb + 1 < nkb) {
      __builtin_prefetch(K + (size_t)(k0 + 32) * DD + lane * 64, 0, 0);
      __builtin_prefetch(V + (size_t)(lane * 2) * TT + (k0 + 32), 0, 0);
    }

    // S = Q @ K^T (two 16x16 tiles over keys k0..k0+31)
    v8f s0 = {}, s1 = {};
    s0 = wmma16(qf0, load_frag_b(K + (size_t)k0 * DD, DD, lane), s0);
    s0 = wmma16(qf1, load_frag_b(K + (size_t)k0 * DD + 32, DD, lane), s0);
    s1 = wmma16(qf0, load_frag_b(K + (size_t)(k0 + 16) * DD, DD, lane), s1);
    s1 = wmma16(qf1, load_frag_b(K + (size_t)(k0 + 16) * DD + 32, DD, lane), s1);

    // V fragments are independent of the softmax: issue their loads now so
    // the latency overlaps the softmax VALU work below.
    v16h vf[4];
#pragma unroll
    for (int s = 0; s < 4; ++s)
      vf[s] = load_frag_b(V + (size_t)(s * 16) * TT + k0, TT, lane);

    // causal mask (scale already folded into Q)
#pragma unroll
    for (int v = 0; v < 8; ++v) {
      const int row = q0 + v + 8 * half;
      if (k0 + n > row)      s0[v] = -__builtin_inff();
      if (k0 + 16 + n > row) s1[v] = -__builtin_inff();
    }

    // online softmax; rows live across the 16 lanes sharing `half`
    float alpha[8];
#pragma unroll
    for (int v = 0; v < 8; ++v) {
      float r = fmaxf(s0[v], s1[v]);
#pragma unroll
      for (int off = 1; off < 16; off <<= 1) r = fmaxf(r, __shfl_xor(r, off, 32));
      const float mnew = fmaxf(mrow[v], r);
      alpha[v] = __expf(mrow[v] - mnew);
      mrow[v] = mnew;
    }
#pragma unroll
    for (int v = 0; v < 8; ++v) {
      const float p0 = __expf(s0[v] - mrow[v]);
      const float p1 = __expf(s1[v] - mrow[v]);
      float rs = p0 + p1;
#pragma unroll
      for (int off = 1; off < 16; off <<= 1) rs += __shfl_xor(rs, off, 32);
      lrow[v] = alpha[v] * lrow[v] + rs;
      const int row16 = v + 8 * half;
      plds[row16 * 32 + n]      = (_Float16)p0;   // P tile -> LDS (A layout source)
      plds[row16 * 32 + 16 + n] = (_Float16)p1;
#pragma unroll
      for (int s = 0; s < 4; ++s) o[s][v] *= alpha[v];
    }
    // wave-local LDS fence: DS ops are in-order per wave; stop compiler reordering
    asm volatile("s_wait_dscnt 0" ::: "memory");

    const v16h pf = load_frag_a(plds, 32, lane);   // P as A fragment (16x32)
#pragma unroll
    for (int s = 0; s < 4; ++s) o[s] = wmma16(pf, vf[s], o[s]);  // O += P @ V
  }

  // normalize + store y as f16 [B,T,C] for the projection GEMM
#pragma unroll
  for (int v = 0; v < 8; ++v) {
    const float il = 1.0f / lrow[v];
    const int t = q0 + v + 8 * half;
#pragma unroll
    for (int s = 0; s < 4; ++s) {
      const int c = h * DD + s * 16 + n;
      yh[((size_t)b * TT + t) * CC + c] = (_Float16)(o[s][v] * il);
    }
  }
}

// ---- output projection: y[4096,1024]f16 @ w_proj + b_proj -> f32 out -------
__global__ __launch_bounds__(256) void gemm_proj(const _Float16* __restrict__ Ah,
                                                 const _Float16* __restrict__ Bt,
                                                 const float* __restrict__ bias,
                                                 float* __restrict__ out) {
  const int lane = threadIdx.x & 31;
  const int gw = blockIdx.x * 8 + (threadIdx.x >> 5);
  const int nt = gw % (CC / 64);           // 16 N-groups of 64
  const int mt = gw / (CC / 64);           // 128 M-tiles of 32
  const int m0 = mt * 32, n0 = nt * 64;
  const int n = lane & 15, half = lane >> 4;

  v8f acc[2][4];
#pragma unroll
  for (int s = 0; s < 4; ++s) {
    float bv = bias[n0 + s * 16 + n];
#pragma unroll
    for (int v = 0; v < 8; ++v) { acc[0][s][v] = bv; acc[1][s][v] = bv; }
  }

  const _Float16* Ap0 = Ah + (size_t)m0 * CC;
  const _Float16* Ap1 = Ap0 + (size_t)16 * CC;
  for (int k = 0; k < CC; k += 32) {
    v16h af0 = load_frag_a(Ap0 + k, CC, lane);
    v16h af1 = load_frag_a(Ap1 + k, CC, lane);
    v16h bf[4];
#pragma unroll
    for (int s = 0; s < 4; ++s)
      bf[s] = load_frag_b(Bt + (size_t)(n0 + s * 16) * CC + k, CC, lane);
#pragma unroll
    for (int s = 0; s < 4; ++s) acc[0][s] = wmma16(af0, bf[s], acc[0][s]);
#pragma unroll
    for (int s = 0; s < 4; ++s) acc[1][s] = wmma16(af1, bf[s], acc[1][s]);
  }

#pragma unroll
  for (int g = 0; g < 2; ++g) {
#pragma unroll
    for (int s = 0; s < 4; ++s) {
      const int c = n0 + s * 16 + n;
      float* dst = out + (size_t)(m0 + g * 16) * CC + c;
#pragma unroll
      for (int v = 0; v < 8; ++v)
        dst[(size_t)(v + 8 * half) * CC] = acc[g][s][v];
    }
  }
}

// ---------------------------------------------------------------------------
extern "C" void kernel_launch(void* const* d_in, const int* in_sizes, int n_in,
                              void* d_out, int out_size, void* d_ws, size_t ws_size,
                              hipStream_t stream) {
  const float* x      = (const float*)d_in[0];   // [2,2048,1024]
  const float* w_attn = (const float*)d_in[1];   // [1024,3072]
  const float* b_attn = (const float*)d_in[2];   // [3072]
  const float* w_proj = (const float*)d_in[3];   // [1024,1024]
  const float* b_proj = (const float*)d_in[4];   // [1024]
  float* out = (float*)d_out;

  const size_t MROWS = (size_t)BB * TT;          // 4096
  char* ws = (char*)d_ws;
  size_t off = 0;
  _Float16* xh  = (_Float16*)(ws + off); off += MROWS * CC * 2;          // 8 MB
  _Float16* wat = (_Float16*)(ws + off); off += (size_t)3 * CC * CC * 2; // 6 MB
  _Float16* wpt = (_Float16*)(ws + off); off += (size_t)CC * CC * 2;     // 2 MB
  _Float16* qh  = (_Float16*)(ws + off); off += MROWS * CC * 2;          // 8 MB
  _Float16* kh  = (_Float16*)(ws + off); off += MROWS * CC * 2;          // 8 MB
  _Float16* vt  = (_Float16*)(ws + off); off += MROWS * CC * 2;          // 8 MB
  _Float16* yh  = (_Float16*)(ws + off); off += MROWS * CC * 2;          // 8 MB

  const int nx = (int)(MROWS * CC);
  cvt_f32_to_f16<<<(nx + 255) / 256, 256, 0, stream>>>(x, xh, nx);
  cvt_transpose<<<dim3(3 * CC / 16, CC / 16), 256, 0, stream>>>(w_attn, wat, CC, 3 * CC);
  cvt_transpose<<<dim3(CC / 16, CC / 16), 256, 0, stream>>>(w_proj, wpt, CC, CC);

  // 128 M-tiles * 48 N-groups = 6144 waves / 8 per block
  gemm_qkv<<<768, 256, 0, stream>>>(xh, wat, b_attn, qh, kh, vt);
  // 2*16*128 q-tiles = 4096 waves / 8 per block
  attn_kernel<<<512, 256, 0, stream>>>(qh, kh, vt, yh);
  // 128 M-tiles * 16 N-groups = 2048 waves / 8 per block
  gemm_proj<<<256, 256, 0, stream>>>(yh, wpt, b_proj, out);
}
